// AutoCorrelation_79336635892451
// MI455X (gfx1250) — matrix operand order repaired and checked by reading, not compiled, not run on hardware
//
#include <hip/hip_runtime.h>
#include <hip/hip_bf16.h>

typedef __attribute__((ext_vector_type(16))) _Float16 v16h;
typedef __attribute__((ext_vector_type(8)))  float    v8f;

#define NB 32
#define LT 512            // sequence length L
#define NC 512            // channels C = H*E
#define LMASK (LT - 1)

#define AS1 __attribute__((address_space(1)))
#define AS3 __attribute__((address_space(3)))

#if defined(__has_builtin)
#  if __has_builtin(__builtin_amdgcn_global_load_async_to_lds_b32) && \
      __has_builtin(__builtin_amdgcn_s_wait_asynccnt)
#    define USE_ASYNC_LDS 1
#  endif
#endif
#ifndef USE_ASYNC_LDS
#  define USE_ASYNC_LDS 0
#endif

__device__ inline void cvt4(v16h& d, int o, float4 f) {
  d[o + 0] = (_Float16)f.x;
  d[o + 1] = (_Float16)f.y;
  d[o + 2] = (_Float16)f.z;
  d[o + 3] = (_Float16)f.w;
}

// ---------------------------------------------------------------------------
// Kernel 1: mean_value[b,tau] = (1/C) * sum_t sum_c K[b,t,c] * Q[b,(t+tau)%L,c]
// Per-wave 16(t) x 32(u) tile of S = K * Q^T: one shared A fragment (K rows),
// two B fragments (Q rows) -> two v_wmma per K-step. Diagonals folded into an
// LDS accumulator, then one global atomic pass per block.
// 512 tiles per batch; 8 waves per block -> 2048 blocks.
// ---------------------------------------------------------------------------
__global__ __launch_bounds__(256) void corr_mean_kernel(
    const float* __restrict__ q, const float* __restrict__ k,
    float* __restrict__ mean_value) {
  __shared__ float lds_mean[LT];
  const int tid  = threadIdx.x;
  const int wave = tid >> 5;
  const int lane = tid & 31;

  lds_mean[tid]       = 0.0f;
  lds_mean[tid + 256] = 0.0f;
  __syncthreads();

  const int tile = blockIdx.x * 8 + wave;   // 512 tiles per batch, 512 % 8 == 0
  const int b    = tile >> 9;
  const int r    = tile & 511;
  const int t0   = (r >> 4) << 4;           // K-rows (time t), 32 tiles
  const int u0   = (r & 15) << 5;           // Q-rows (time u = t + tau), 16 pair-tiles

  const float* kb = k + (size_t)b * LT * NC;
  const float* qb = q + (size_t)b * LT * NC;

  const int hi    = (lane < 16) ? 0 : 1;    // which half-wave
  const int lane16 = lane & 15;

  const float* arow = kb + (size_t)(t0 + lane16) * NC + (hi ? 16 : 0);
  const float* brow0 = qb + (size_t)(u0 + lane16) * NC + (hi ? 16 : 0);
  const float* brow1 = brow0 + (size_t)16 * NC;

  v8f acc0 = {}, acc1 = {};
  for (int kk = 0; kk < NC; kk += 32) {
    const float* ap  = arow  + kk;
    const float* bp0 = brow0 + kk;
    const float* bp1 = brow1 + kk;
    __builtin_prefetch(ap + 32, 0, 1);
    __builtin_prefetch(bp0 + 32, 0, 1);
    __builtin_prefetch(bp1 + 32, 0, 1);

    v16h af, bf0, bf1;
    // A: lane<16 cols kk..kk+15; lane>=16 cols kk+16..kk+31
    cvt4(af, 0,  *(const float4*)(ap + 0));
    cvt4(af, 4,  *(const float4*)(ap + 4));
    cvt4(af, 8,  *(const float4*)(ap + 8));
    cvt4(af, 12, *(const float4*)(ap + 12));
    // B: lane n holds column n, K = kk..kk+15 (lanes<16) / kk+16..kk+31
    cvt4(bf0, 0,  *(const float4*)(bp0 + 0));
    cvt4(bf0, 4,  *(const float4*)(bp0 + 4));
    cvt4(bf0, 8,  *(const float4*)(bp0 + 8));
    cvt4(bf0, 12, *(const float4*)(bp0 + 12));
    cvt4(bf1, 0,  *(const float4*)(bp1 + 0));
    cvt4(bf1, 4,  *(const float4*)(bp1 + 4));
    cvt4(bf1, 8,  *(const float4*)(bp1 + 8));
    cvt4(bf1, 12, *(const float4*)(bp1 + 12));

    acc0 = __builtin_amdgcn_wmma_f32_16x16x32_f16(false, af, false, bf0,
                                                  (short)0, acc0, false, false);
    acc1 = __builtin_amdgcn_wmma_f32_16x16x32_f16(false, af, false, bf1,
                                                  (short)0, acc1, false, false);
  }

  // Fold S[t0+m, u+n] into diagonal tau = (u+n - t0-m) mod L, scaled by 1/C.
  const float scale = 1.0f / (float)NC;
  const int   mb    = hi ? 8 : 0;
#pragma unroll
  for (int rr = 0; rr < 8; ++rr) {
    const int m    = rr + mb;
    const int tau0 = (u0 + lane16 - t0 - m + LT) & LMASK;
    const int tau1 = (u0 + 16 + lane16 - t0 - m + LT) & LMASK;
    atomicAdd(&lds_mean[tau0], acc0[rr] * scale);
    atomicAdd(&lds_mean[tau1], acc1[rr] * scale);
  }
  __syncthreads();
  atomicAdd(&mean_value[b * LT + tid],       lds_mean[tid]);
  atomicAdd(&mean_value[b * LT + tid + 256], lds_mean[tid + 256]);
}

// ---------------------------------------------------------------------------
// Kernel 2: per-batch descending bitonic sort (keys + argsort indices) of
// mean_value, softmax over sorted keys -> probs, indices -> delays.
// One block (256 threads) per batch, 512-element sort in LDS.
// ---------------------------------------------------------------------------
__global__ __launch_bounds__(256) void sort_softmax_kernel(
    const float* __restrict__ mean_value,
    float* __restrict__ probs, int* __restrict__ delays) {
  __shared__ float key[LT];
  __shared__ int   idx[LT];
  __shared__ float Zs;
  const int b   = blockIdx.x;
  const int tid = threadIdx.x;

  for (int i = tid; i < LT; i += 256) { key[i] = mean_value[b * LT + i]; idx[i] = i; }
  __syncthreads();

  for (int kk = 2; kk <= LT; kk <<= 1) {
    for (int j = kk >> 1; j > 0; j >>= 1) {
      for (int i = tid; i < LT; i += 256) {
        const int ixj = i ^ j;
        if (ixj > i) {
          const bool  desc = ((i & kk) == 0);
          const float ki = key[i], kj = key[ixj];
          if ((ki < kj) == desc) {
            key[i] = kj; key[ixj] = ki;
            const int t = idx[i]; idx[i] = idx[ixj]; idx[ixj] = t;
          }
        }
      }
      __syncthreads();
    }
  }

  if (tid == 0) Zs = 0.0f;
  __syncthreads();
  const float mx = key[0];                          // max after descending sort
  const float e0 = __expf(key[tid]       - mx);
  const float e1 = __expf(key[tid + 256] - mx);
  atomicAdd(&Zs, e0 + e1);
  __syncthreads();
  const float inv = 1.0f / Zs;
  probs[b * LT + tid]        = e0 * inv;
  probs[b * LT + tid + 256]  = e1 * inv;
  delays[b * LT + tid]       = idx[tid];
  delays[b * LT + tid + 256] = idx[tid + 256];
}

// ---------------------------------------------------------------------------
// Kernel 2b: g[b, delays0[k]] = probs[b, k]  (delays from batch 0 only)
// ---------------------------------------------------------------------------
__global__ void scatter_g_kernel(const float* __restrict__ probs,
                                 const int* __restrict__ delays,
                                 float* __restrict__ g) {
  const int b = blockIdx.x, tid = threadIdx.x;
  for (int i = tid; i < LT; i += 256)
    g[b * LT + delays[i]] = probs[b * LT + i];     // delays row 0
}

// ---------------------------------------------------------------------------
// Kernel 3: out[b,t,c] = sum_u g[b,(u-t)%L] * V[b,u,c]  == G_b (circulant) x V_b
// Per-wave 16(t) x 32(c) tile: one circulant A fragment gathered from LDS,
// two V B-fragments -> two v_wmma per u-step. g staged into LDS with
// GLOBAL_LOAD_ASYNC_TO_LDS when available (ASYNCcnt path).
// ---------------------------------------------------------------------------
__global__ __launch_bounds__(256) void agg_kernel(
    const float* __restrict__ v, const float* __restrict__ g,
    float* __restrict__ out) {
  __shared__ float gs[LT];
  const int tid  = threadIdx.x;
  const int wave = tid >> 5;
  const int lane = tid & 31;

  const int tile = blockIdx.x * 8 + wave;   // 512 tiles per batch
  const int b    = tile >> 9;
  const int r    = tile & 511;
  const int t0   = (r >> 4) << 4;           // output time tile
  const int c0   = (r & 15) << 5;           // output channel pair-tile

#if USE_ASYNC_LDS
  __builtin_amdgcn_global_load_async_to_lds_b32(
      (AS1 int*)(g + (size_t)b * LT + tid),       (AS3 int*)&gs[tid],       0, 0);
  __builtin_amdgcn_global_load_async_to_lds_b32(
      (AS1 int*)(g + (size_t)b * LT + tid + 256), (AS3 int*)&gs[tid + 256], 0, 0);
  __builtin_amdgcn_s_wait_asynccnt(0);
#else
  gs[tid]       = g[b * LT + tid];
  gs[tid + 256] = g[b * LT + tid + 256];
#endif
  __syncthreads();

  const float* vb = v + (size_t)b * LT * NC;
  const int hi     = (lane < 16) ? 0 : 1;
  const int lane16 = lane & 15;

  v8f acc0 = {}, acc1 = {};
  for (int kk = 0; kk < LT; kk += 32) {
    v16h af, bf0, bf1;
    // A[m, col] = g[(kk+col - t0 - m) mod L]; col sets per half-wave as in k1
    const int cb   = hi ? 16 : 0;
    const int base = kk + cb - t0 - lane16 + 2 * LT;   // keep index positive
#pragma unroll
    for (int j = 0; j < 8; ++j)  af[j]     = (_Float16)gs[(base + j) & LMASK];
#pragma unroll
    for (int j = 0; j < 8; ++j)  af[8 + j] = (_Float16)gs[(base + 8 + j) & LMASK];
    // B[kl, n] = V[b, kk + koff + kl, c0(+16) + n]  (columns of V)
    const float* bp0 = vb + (size_t)(kk + (hi ? 16 : 0)) * NC + c0 + lane16;
    const float* bp1 = bp0 + 16;
#pragma unroll
    for (int j = 0; j < 16; ++j) bf0[j] = (_Float16)bp0[(size_t)j * NC];
#pragma unroll
    for (int j = 0; j < 16; ++j) bf1[j] = (_Float16)bp1[(size_t)j * NC];

    acc0 = __builtin_amdgcn_wmma_f32_16x16x32_f16(false, af, false, bf0,
                                                  (short)0, acc0, false, false);
    acc1 = __builtin_amdgcn_wmma_f32_16x16x32_f16(false, af, false, bf1,
                                                  (short)0, acc1, false, false);
  }

  float* ob = out + (size_t)b * LT * NC;
#pragma unroll
  for (int rr = 0; rr < 8; ++rr) {
    const int m = rr + (hi ? 8 : 0);
    ob[(size_t)(t0 + m) * NC + c0 + lane16]      = acc0[rr];
    ob[(size_t)(t0 + m) * NC + c0 + 16 + lane16] = acc1[rr];
  }
}

// ---------------------------------------------------------------------------
extern "C" void kernel_launch(void* const* d_in, const int* in_sizes, int n_in,
                              void* d_out, int out_size, void* d_ws, size_t ws_size,
                              hipStream_t stream) {
  const float* q = (const float*)d_in[0];
  const float* k = (const float*)d_in[1];
  const float* v = (const float*)d_in[2];
  float* out = (float*)d_out;

  char*  ws         = (char*)d_ws;
  float* mean_value = (float*)(ws);                 //  64 KB  [32,512] f32
  float* probs      = (float*)(ws + (64 << 10));    //  64 KB  [32,512] f32
  int*   delays     = (int*)  (ws + (128 << 10));   //  64 KB  [32,512] i32
  float* gbuf       = (float*)(ws + (192 << 10));   //  64 KB  [32,512] f32

  (void)hipMemsetAsync(mean_value, 0, NB * LT * sizeof(float), stream);

  // 32 batches * 512 tiles / 8 waves = 2048 blocks
  corr_mean_kernel<<<2048, 256, 0, stream>>>(q, k, mean_value);
  sort_softmax_kernel<<<NB, 256, 0, stream>>>(mean_value, probs, delays);
  scatter_g_kernel<<<NB, 256, 0, stream>>>(probs, delays, gbuf);
  agg_kernel<<<2048, 256, 0, stream>>>(v, gbuf, out);
}
